// RRec_12927851561602
// MI455X (gfx1250) — matrix operand order repaired and checked by reading, not compiled, not run on hardware
//
#include <hip/hip_runtime.h>
#include <hip/hip_bf16.h>

// ---------------------------------------------------------------------------
// CDNA5 (gfx1250, wave32) implementation of the SASRec+MoE forward.
// All dense matmuls go through a bf16 WMMA (V_WMMA_F32_16X16X32_BF16) tile
// kernel (16x64 per wave: one A fragment reused by 4 WMMAs per K-step).
// The K-loop is software-pipelined: the 20 float4 loads of step k+1 are
// issued before the converts/WMMAs of step k and consumed only in the next
// iteration, so they stay in flight behind the matrix work.
// Attention uses an online-softmax wave32 kernel; LN/gating use LDS
// reductions.  Fragments use contiguous k = 16*half + s in both A and B.
// ---------------------------------------------------------------------------

typedef __attribute__((ext_vector_type(16))) __bf16 v16bf;
typedef __attribute__((ext_vector_type(8)))  float  v8f;

#define EPS 1e-8f

// Pack two float4 (8 contiguous fp32) into dst[o..o+7] as bf16.
__device__ __forceinline__ void pack8(v16bf& dst, int o, float4 a, float4 b)
{
    dst[o + 0] = (__bf16)a.x;  dst[o + 1] = (__bf16)a.y;
    dst[o + 2] = (__bf16)a.z;  dst[o + 3] = (__bf16)a.w;
    dst[o + 4] = (__bf16)b.x;  dst[o + 5] = (__bf16)b.y;
    dst[o + 6] = (__bf16)b.z;  dst[o + 7] = (__bf16)b.w;
}

// ---------------------------------------------------------------------------
// Generic GEMM: C[M,N] = A[M,K] @ W[N,K]^T (+ bias[N]) (+ optional relu)
// A fp32 row-major (row stride lda), W fp32 row-major (row stride ldw),
// C fp32 (row stride ldc).  One wave32 per 16x64 output block (4 x 16x16
// WMMA tiles sharing the A fragment); K in steps of 32 through
// v_wmma_f32_16x16x32_bf16.  K must be a multiple of 32; lda/ldw multiples
// of 4 (float4 loads).  Rows/cols clamped for loads, guarded for stores
// (handles N=20001 / M=20001).
// amdgpu_waves_per_eu(1): allow a large per-wave VGPR budget so the
// double-buffered staging quads stay in distinct registers.
// ---------------------------------------------------------------------------
__global__ __attribute__((amdgpu_waves_per_eu(1))) __launch_bounds__(32)
void gemm_bf16_wmma_kernel(const float* __restrict__ A, int lda,
                           const float* __restrict__ W, int ldw,
                           const float* __restrict__ bias,
                           float* __restrict__ C, long ldc,
                           int M, int Nn, int K, int act)
{
    const int lane = threadIdx.x;          // 0..31
    const int half = lane >> 4;            // 0 or 1
    const int l16  = lane & 15;

    const int m  = blockIdx.x * 16 + l16;  // A row handled by this lane (frag A)
    const int mc = (m < M) ? m : (M - 1);
    const float* Arow = A + (long)mc * lda;

    const int n0 = blockIdx.y * 64;        // first output column of this block
    int nn[4];
    const float* Wrow[4];
    #pragma unroll
    for (int t = 0; t < 4; ++t) {
        nn[t] = n0 + t * 16 + l16;         // output column / W row (frag B)
        const int nc = (nn[t] < Nn) ? nn[t] : (Nn - 1);
        Wrow[t] = W + (long)nc * ldw;
    }

    v8f acc[4];
    #pragma unroll
    for (int t = 0; t < 4; ++t)
        acc[t] = (v8f){0.f, 0.f, 0.f, 0.f, 0.f, 0.f, 0.f, 0.f};

    // ---- prologue: load K-step 0 ----
    // A 16x32 bf16 fragment (row per lane): contiguous 16-float run at
    // K = kk + 16*half         (VGPR j: kb = 16*half + 2j)
    // B 32x16 bf16 fragment (column per lane): contiguous 16-float run at
    // K = kk + 16*half         (VGPR j: kb = 16*half + 2j)
    float4 ca[4], cb[4][4];
    {
        const float* ap = Arow + (half << 4);
        ca[0] = *(const float4*)(ap);
        ca[1] = *(const float4*)(ap + 4);
        ca[2] = *(const float4*)(ap + 8);
        ca[3] = *(const float4*)(ap + 12);
        #pragma unroll
        for (int t = 0; t < 4; ++t) {
            const float4* p = (const float4*)(Wrow[t] + (half << 4));
            cb[t][0] = p[0];  cb[t][1] = p[1];
            cb[t][2] = p[2];  cb[t][3] = p[3];
        }
    }

    #pragma unroll 2
    for (int kk = 0; kk < K; kk += 32) {
        const int kn = kk + 32;
        // ---- issue next step's 20 b128 loads before this step's math ----
        float4 na[4], nb[4][4];
        if (kn < K) {
            const float* ap = Arow + kn + (half << 4);
            na[0] = *(const float4*)(ap);
            na[1] = *(const float4*)(ap + 4);
            na[2] = *(const float4*)(ap + 8);
            na[3] = *(const float4*)(ap + 12);
            #pragma unroll
            for (int t = 0; t < 4; ++t) {
                const float4* p = (const float4*)(Wrow[t] + kn + (half << 4));
                nb[t][0] = p[0];  nb[t][1] = p[1];
                nb[t][2] = p[2];  nb[t][3] = p[3];
            }
        }
        // ---- convert + 4 WMMAs sharing the A fragment (current step) ----
        v16bf af;
        pack8(af, 0, ca[0], ca[1]);
        pack8(af, 8, ca[2], ca[3]);
        #pragma unroll
        for (int t = 0; t < 4; ++t) {
            v16bf bfm;
            pack8(bfm, 0, cb[t][0], cb[t][1]);
            pack8(bfm, 8, cb[t][2], cb[t][3]);
            acc[t] = __builtin_amdgcn_wmma_f32_16x16x32_bf16(
                         false, af, false, bfm, (short)0, acc[t], false, false);
        }
        // ---- rotate buffers (elided by unroll-2 SSA renaming) ----
        if (kn < K) {
            #pragma unroll
            for (int i = 0; i < 4; ++i) ca[i] = na[i];
            #pragma unroll
            for (int t = 0; t < 4; ++t) {
                #pragma unroll
                for (int i = 0; i < 4; ++i) cb[t][i] = nb[t][i];
            }
        }
    }

    // C/D layout: VGPR i -> row = 16*tileM + i + 8*half, col = nn[t]
    const int r0 = blockIdx.x * 16 + (half << 3);
    const bool full = (blockIdx.x * 16 + 16 <= M) && (n0 + 64 <= Nn);
    if (full) {
        #pragma unroll
        for (int t = 0; t < 4; ++t) {
            const float bv = bias ? bias[nn[t]] : 0.0f;
            #pragma unroll
            for (int i = 0; i < 8; ++i) {
                float v = acc[t][i] + bv;
                if (act) v = fmaxf(v, 0.0f);
                C[(long)(r0 + i) * ldc + nn[t]] = v;
            }
        }
    } else {
        #pragma unroll
        for (int t = 0; t < 4; ++t) {
            const float bv = bias ? bias[(nn[t] < Nn) ? nn[t] : (Nn - 1)] : 0.0f;
            #pragma unroll
            for (int i = 0; i < 8; ++i) {
                float v = acc[t][i] + bv;
                if (act) v = fmaxf(v, 0.0f);
                if (r0 + i < M && nn[t] < Nn) C[(long)(r0 + i) * ldc + nn[t]] = v;
            }
        }
    }
}

// ---------------------------------------------------------------------------
// Online-softmax attention, dh = 64 fixed.  One wave32 per (b,h,e,l) query.
// score = scale * (q . k_j) + causal mask (j <= l).  Lane holds elems
// {lane, lane+32}; dot reduced with wave32 shfl_xor.
// ---------------------------------------------------------------------------
__global__ __launch_bounds__(32)
void attn_kernel(const float* __restrict__ q, const float* __restrict__ k,
                 const float* __restrict__ v, float* __restrict__ o,
                 long qsB, long qsH, long qsE, long qsL,
                 long ksB, long ksH, long ksL,
                 long osB, long osH, long osE, long osL,
                 int Hh, int Ee, int L, float scale)
{
    const int lane = threadIdx.x;
    long idx = blockIdx.x;
    const int l = (int)(idx % L); idx /= L;
    const int e = (int)(idx % Ee); idx /= Ee;
    const int h = (int)(idx % Hh);
    const int b = (int)(idx / Hh);

    const float* qp = q + (long)b * qsB + (long)h * qsH + (long)e * qsE + (long)l * qsL;
    const float q0 = qp[lane];
    const float q1 = qp[lane + 32];
    const float* kbase = k + (long)b * ksB + (long)h * ksH;
    const float* vbase = v + (long)b * ksB + (long)h * ksH;

    float mx = -3.4e38f, den = 0.f, a0 = 0.f, a1 = 0.f;
    for (int j = 0; j <= l; ++j) {
        const float* kp = kbase + (long)j * ksL;
        float p = q0 * kp[lane] + q1 * kp[lane + 32];
        for (int off = 16; off; off >>= 1) p += __shfl_xor(p, off, 32);
        const float s  = p * scale;
        const float nm = fmaxf(mx, s);
        const float corr = __expf(mx - nm);   // 0 on first iteration
        const float w    = __expf(s - nm);
        const float* vp = vbase + (long)j * ksL;
        a0 = a0 * corr + w * vp[lane];
        a1 = a1 * corr + w * vp[lane + 32];
        den = den * corr + w;
        mx = nm;
    }
    const float inv = 1.0f / den;
    float* op = o + (long)b * osB + (long)h * osH + (long)e * osE + (long)l * osL;
    op[lane]      = a0 * inv;
    op[lane + 32] = a1 * inv;
}

// ---------------------------------------------------------------------------
// Row LayerNorm: y = (x-mu)/sqrt(var+eps)*g+b.  One 64-thread block per row.
// W must be 64 or 256.  In-place safe (reads cached in regs before writes).
// ---------------------------------------------------------------------------
__global__ __launch_bounds__(64)
void ln_kernel(const float* __restrict__ x, const float* __restrict__ g,
               const float* __restrict__ bt, float* __restrict__ y, int W)
{
    __shared__ float red[64];
    const int t = threadIdx.x;
    const long row = blockIdx.x;
    const float* xr = x + row * (long)W;
    const int per = W >> 6;                 // 1 or 4
    float v[4];
    float s = 0.f;
    for (int i = 0; i < per; ++i) { v[i] = xr[t + (i << 6)]; s += v[i]; }
    red[t] = s; __syncthreads();
    for (int off = 32; off; off >>= 1) { if (t < off) red[t] += red[t + off]; __syncthreads(); }
    const float mu = red[0] / (float)W;  __syncthreads();
    float s2 = 0.f;
    for (int i = 0; i < per; ++i) { const float d = v[i] - mu; s2 += d * d; }
    red[t] = s2; __syncthreads();
    for (int off = 32; off; off >>= 1) { if (t < off) red[t] += red[t + off]; __syncthreads(); }
    const float inv = rsqrtf(red[0] / (float)W + EPS);
    float* yr = y + row * (long)W;
    for (int i = 0; i < per; ++i)
        yr[t + (i << 6)] = (v[i] - mu) * inv * g[t + (i << 6)] + bt[t + (i << 6)];
}

// out[i] = a[i] + b[i]
__global__ __launch_bounds__(256)
void add_kernel(float* __restrict__ out, const float* __restrict__ a,
                const float* __restrict__ b, long n)
{
    const long i = (long)blockIdx.x * 256 + threadIdx.x;
    if (i < n) out[i] = a[i] + b[i];
}

// seqs[b,l,:] = item_emb[id] + pos_emb[(id!=0)?(l+1):0]
__global__ __launch_bounds__(256)
void embed_kernel(const float* __restrict__ ie, const float* __restrict__ pe,
                  const int* __restrict__ seq, float* __restrict__ out, int L)
{
    const int t = threadIdx.x;
    const long idx = blockIdx.x;            // b*L + l
    const int l = (int)(idx % L);
    const int id = seq[idx];
    const int p = (id != 0) ? (l + 1) : 0;
    out[idx * 256 + t] = ie[(long)id * 256 + t] + pe[(long)p * 256 + t];
}

// o[b,h,e,l,u] = 0 where sx[b,l,h,u] == 0   (pad-mask quirk)
__global__ __launch_bounds__(256)
void padmask_kernel(float* __restrict__ o, const float* __restrict__ sx,
                    int Hh, int Ee, int L, long n)
{
    const long i = (long)blockIdx.x * 256 + threadIdx.x;
    if (i >= n) return;
    const int u = (int)(i & 63);
    long r = i >> 6;
    const int l = (int)(r % L); r /= L;
    const int e = (int)(r % Ee); r /= Ee; (void)e;
    const int h = (int)(r % Hh);
    const int b = (int)(r / Hh);
    const float sv = sx[((((long)b * L + l) * Hh + h) << 6) + u];
    if (sv == 0.f) o[i] = 0.f;
}

// MoE gating + expert combine + diag residual + LN(64) -> f_n[b,l,h,u]
__global__ __launch_bounds__(64)
void moe_gate_kernel(const float* __restrict__ o, const float* __restrict__ sx,
                     const float* __restrict__ gw, const float* __restrict__ gb,
                     const float* __restrict__ lng, const float* __restrict__ lnb,
                     float* __restrict__ fn, int Hh, int Ee, int L)
{
    __shared__ float red[64];
    __shared__ float gsh[4];
    const int t = threadIdx.x;
    long idx = blockIdx.x;
    const int l = (int)(idx % L); idx /= L;
    const int h = (int)(idx % Hh);
    const int b = (int)(idx / Hh);

    float ov[4];
    for (int e = 0; e < 4; ++e)
        ov[e] = o[((((long)(b * Hh + h) * Ee + e) * L + l) << 6) + t];

    // logits over concat of experts: gw[h,eo, e*64+u]
    for (int eo = 0; eo < 4; ++eo) {
        float p = 0.f;
        for (int e = 0; e < 4; ++e)
            p += ov[e] * gw[((long)(h * Ee + eo)) * (Ee * 64) + (e << 6) + t];
        red[t] = p; __syncthreads();
        for (int off = 32; off; off >>= 1) { if (t < off) red[t] += red[t + off]; __syncthreads(); }
        if (t == 0) gsh[eo] = red[0] + gb[h * Ee + eo];
        __syncthreads();
    }
    if (t == 0) {
        const float m = fmaxf(fmaxf(gsh[0], gsh[1]), fmaxf(gsh[2], gsh[3]));
        float s = 0.f;
        for (int e = 0; e < 4; ++e) { gsh[e] = __expf(gsh[e] - m); s += gsh[e]; }
        for (int e = 0; e < 4; ++e) gsh[e] /= s;
    }
    __syncthreads();
    float comb = 0.f;
    for (int e = 0; e < 4; ++e) comb += gsh[e] * ov[e];
    // residual quirk: batch-row h of head h
    const float dg = sx[((((long)h * L + l) * Hh + h) << 6) + t];
    const float val = comb + dg;
    // LN over 64
    red[t] = val; __syncthreads();
    for (int off = 32; off; off >>= 1) { if (t < off) red[t] += red[t + off]; __syncthreads(); }
    const float mu = red[0] * (1.f / 64.f); __syncthreads();
    const float d = val - mu;
    red[t] = d * d; __syncthreads();
    for (int off = 32; off; off >>= 1) { if (t < off) red[t] += red[t + off]; __syncthreads(); }
    const float inv = rsqrtf(red[0] * (1.f / 64.f) + EPS);
    fn[((((long)b * L + l) * Hh + h) << 6) + t] = d * inv * lng[t] + lnb[t];
}

// h1 = (ch==0) ? 0 : relu(h1)   (in-place on h1)
__global__ __launch_bounds__(256)
void ffn_act_kernel(float* __restrict__ h1, const float* __restrict__ ch, long n)
{
    const long i = (long)blockIdx.x * 256 + threadIdx.x;
    if (i < n) {
        const float c = ch[i];
        h1[i] = (c == 0.f) ? 0.f : fmaxf(h1[i], 0.f);
    }
}

// y = LN( ((ch==0)?0:h2) + ch ) over 64, in-place-safe vs ch
__global__ __launch_bounds__(64)
void ffn_ln_kernel(const float* __restrict__ ch, const float* __restrict__ h2,
                   const float* __restrict__ g, const float* __restrict__ bt,
                   float* __restrict__ y)
{
    __shared__ float red[64];
    const int t = threadIdx.x;
    const long row = blockIdx.x;
    const float c  = ch[(row << 6) + t];
    const float hv = h2[(row << 6) + t];
    const float val = ((c == 0.f) ? 0.f : hv) + c;
    red[t] = val; __syncthreads();
    for (int off = 32; off; off >>= 1) { if (t < off) red[t] += red[t + off]; __syncthreads(); }
    const float mu = red[0] * (1.f / 64.f); __syncthreads();
    const float d = val - mu;
    red[t] = d * d; __syncthreads();
    for (int off = 32; off; off >>= 1) { if (t < off) red[t] += red[t + off]; __syncthreads(); }
    const float inv = rsqrtf(red[0] * (1.f / 64.f) + EPS);
    y[(row << 6) + t] = d * inv * g[t] + bt[t];
}

// gg = softmax(y_row @ ffn_gate_w^T) over H=4; out = y * gg[head]
__global__ __launch_bounds__(64)
void ffn_gate_kernel(const float* __restrict__ y, const float* __restrict__ gw,
                     float* __restrict__ out)
{
    __shared__ float red[64];
    __shared__ float gsh[4];
    const int t = threadIdx.x;
    const long row = blockIdx.x;
    float yv[4];
    for (int i = 0; i < 4; ++i) yv[i] = y[row * 256 + (i << 6) + t];
    for (int h = 0; h < 4; ++h) {
        float p = 0.f;
        for (int i = 0; i < 4; ++i) p += yv[i] * gw[h * 256 + (i << 6) + t];
        red[t] = p; __syncthreads();
        for (int off = 32; off; off >>= 1) { if (t < off) red[t] += red[t + off]; __syncthreads(); }
        if (t == 0) gsh[h] = red[0];
        __syncthreads();
    }
    if (t == 0) {
        const float m = fmaxf(fmaxf(gsh[0], gsh[1]), fmaxf(gsh[2], gsh[3]));
        float s = 0.f;
        for (int h = 0; h < 4; ++h) { gsh[h] = __expf(gsh[h] - m); s += gsh[h]; }
        for (int h = 0; h < 4; ++h) gsh[h] /= s;
    }
    __syncthreads();
    for (int i = 0; i < 4; ++i) out[row * 256 + (i << 6) + t] = yv[i] * gsh[i];
}

// ---------------------------------------------------------------------------
static inline void gemm(const float* A, int lda, const float* W, int ldw,
                        const float* bias, float* C, long ldc,
                        int M, int N, int K, int act, hipStream_t s)
{
    dim3 g((M + 15) / 16, (N + 63) / 64);
    gemm_bf16_wmma_kernel<<<g, dim3(32), 0, s>>>(A, lda, W, ldw, bias, C, ldc, M, N, K, act);
}

extern "C" void kernel_launch(void* const* d_in, const int* in_sizes, int n_in,
                              void* d_out, int out_size, void* d_ws, size_t ws_size,
                              hipStream_t stream)
{
    (void)in_sizes; (void)n_in; (void)out_size; (void)ws_size;
    const int B = 32, L = 200, D = 256, H = 4, E = 4, HU = 64, NH = 4, NB = 2;
    const int NTOT = 20001;
    const long BL = (long)B * L;            // 6400

    // ---- inputs (setup_inputs order; params dict flattened in key order) ----
    const int*   log_seqs   = (const int*)  d_in[1];
    const float* item_emb   = (const float*)d_in[4];
    const float* pos_emb    = (const float*)d_in[5];
    const float* attn_ln_g  = (const float*)d_in[6];
    const float* attn_ln_b  = (const float*)d_in[7];
    const float* in_w       = (const float*)d_in[8];
    const float* in_b       = (const float*)d_in[9];
    const float* out_w      = (const float*)d_in[10];
    const float* out_b      = (const float*)d_in[11];
    const float* fwd_ln_g   = (const float*)d_in[12];
    const float* fwd_ln_b   = (const float*)d_in[13];
    const float* c1_w       = (const float*)d_in[14];
    const float* c1_b       = (const float*)d_in[15];
    const float* c2_w       = (const float*)d_in[16];
    const float* c2_b       = (const float*)d_in[17];
    const float* last_ln_g  = (const float*)d_in[18];
    const float* last_ln_b  = (const float*)d_in[19];
    const float* head_w     = (const float*)d_in[20];
    const float* head_b     = (const float*)d_in[21];
    const float* wq         = (const float*)d_in[22];
    const float* wk         = (const float*)d_in[23];
    const float* wv         = (const float*)d_in[24];
    const float* moe_gate_w = (const float*)d_in[25];
    const float* moe_gate_b = (const float*)d_in[26];
    const float* moe_ln_g   = (const float*)d_in[27];
    const float* moe_ln_b   = (const float*)d_in[28];
    const float* fc_w       = (const float*)d_in[29];
    const float* fc_b       = (const float*)d_in[30];
    const float* fc2_w      = (const float*)d_in[31];
    const float* fc2_b      = (const float*)d_in[32];
    const float* ffn_ln_g   = (const float*)d_in[33];
    const float* ffn_ln_b   = (const float*)d_in[34];
    const float* ffn_gate_w = (const float*)d_in[35];
    const float* lab_w      = (const float*)d_in[36];

    // ---- workspace layout (floats) ----
    float* w    = (float*)d_ws;
    float* seqs = w;                       // [B*L, 256]
    float* Qbuf = seqs + BL * D;           // [B*L, 256]
    float* qkv  = Qbuf + BL * D;           // [B*L, 768]
    float* attn = qkv  + BL * 3 * D;       // [B*L, 256]   (reused as h1)
    float* tmp2 = attn + BL * D;           // [B*L, 256]   (reused as h2)
    float* sx   = tmp2 + BL * D;           // [B,L,H,HU]
    float* qe   = sx   + BL * D;           // [H,E,B*L,HU] (reused as ph)
    float* kb   = qe   + (long)H * E * BL * HU;   // [H,B*L,HU]
    float* vb   = kb   + (long)H * BL * HU;       // [H,B*L,HU]
    float* ob   = vb   + (long)H * BL * HU;       // [B,H,E,L,HU]
    float* fn   = ob   + (long)B * H * E * L * HU; // [B,L,H,HU] (reused as y)

    float* log_feats = (float*)d_out;                    // [B,L,D]
    float* score     = log_feats + BL * D;               // [B,L,NTOT]

    const long nBD = BL * D;

    // ---- embedding + positions ----
    embed_kernel<<<(int)BL, 256, 0, stream>>>(item_emb, pos_emb, log_seqs, seqs, L);

    // ---- SASRec stack: NB models x NB blocks ----
    for (int s = 0; s < NB; ++s) {
        for (int blk = 0; blk < NB; ++blk) {
            const long sb = s * NB + blk;
            // Q = LN(seqs)
            ln_kernel<<<(int)BL, 64, 0, stream>>>(seqs, attn_ln_g + sb * D, attn_ln_b + sb * D, Qbuf, D);
            // q = Q @ in_w[:D]^T + b ; k,v = seqs @ in_w[D:]^T + b
            gemm(Qbuf, D, in_w + sb * 3 * D * D, D, in_b + sb * 3 * D,
                 qkv, 3 * D, (int)BL, D, D, 0, stream);
            gemm(seqs, D, in_w + sb * 3 * D * D + (long)D * D, D, in_b + sb * 3 * D + D,
                 qkv + D, 3 * D, (int)BL, 2 * D, D, 0, stream);
            // causal attention (scale = 1/sqrt(64))
            attn_kernel<<<B * NH * L, 32, 0, stream>>>(
                qkv, qkv + D, qkv + 2 * D, attn,
                (long)L * 768, 64, 0, 768,
                (long)L * 768, 64, 768,
                (long)L * 256, 64, 0, 256,
                NH, 1, L, 0.125f);
            // out projection, residual with Q
            gemm(attn, D, out_w + sb * D * D, D, out_b + sb * D, tmp2, D, (int)BL, D, D, 0, stream);
            add_kernel<<<(int)((nBD + 255) / 256), 256, 0, stream>>>(seqs, Qbuf, tmp2, nBD);
            // fwd LN (in place)
            ln_kernel<<<(int)BL, 64, 0, stream>>>(seqs, fwd_ln_g + sb * D, fwd_ln_b + sb * D, seqs, D);
            // FFN: relu(seqs@c1^T+b)@c2^T+b, residual
            gemm(seqs, D, c1_w + sb * D * D, D, c1_b + sb * D, tmp2, D, (int)BL, D, D, 1, stream);
            gemm(tmp2, D, c2_w + sb * D * D, D, c2_b + sb * D, attn, D, (int)BL, D, D, 0, stream);
            add_kernel<<<(int)((nBD + 255) / 256), 256, 0, stream>>>(seqs, seqs, attn, nBD);
        }
        ln_kernel<<<(int)BL, 64, 0, stream>>>(seqs, last_ln_g + (long)s * D, last_ln_b + (long)s * D, seqs, D);
    }

    // ---- MoE multi-head ----
    // sx[b,l,h,u] = seqs @ head_w^T + head_b   (head_w flat [256,256])
    gemm(seqs, D, head_w, D, head_b, sx, D, (int)BL, D, D, 0, stream);
    // per-(h,e) expert query projections; per-h k,v
    for (int h = 0; h < H; ++h) {
        for (int e = 0; e < E; ++e)
            gemm(sx + h * HU, D, wq + ((long)(h * E + e)) * HU * HU, HU, nullptr,
                 qe + ((long)(h * E + e)) * BL * HU, HU, (int)BL, HU, HU, 0, stream);
        gemm(sx + h * HU, D, wk + (long)h * HU * HU, HU, nullptr,
             kb + (long)h * BL * HU, HU, (int)BL, HU, HU, 0, stream);
        gemm(sx + h * HU, D, wv + (long)h * HU * HU, HU, nullptr,
             vb + (long)h * BL * HU, HU, (int)BL, HU, HU, 0, stream);
    }
    // expert attention (NOTE: reference MULTIPLIES by sqrt(HU) = 8)
    attn_kernel<<<B * H * E * L, 32, 0, stream>>>(
        qe, kb, vb, ob,
        (long)L * HU, (long)E * BL * HU, (long)BL * HU, HU,
        (long)L * HU, (long)BL * HU, HU,
        (long)H * E * L * HU, (long)E * L * HU, (long)L * HU, HU,
        H, E, L, 8.0f);
    // pad-mask quirk: zero o where sx==0
    {
        const long n = (long)B * H * E * L * HU;
        padmask_kernel<<<(int)((n + 255) / 256), 256, 0, stream>>>(ob, sx, H, E, L, n);
    }
    // gating + combine + diag residual + LN -> fn[b,l,h,u]
    moe_gate_kernel<<<B * H * L, 64, 0, stream>>>(ob, sx, moe_gate_w, moe_gate_b,
                                                  moe_ln_g, moe_ln_b, fn, H, E, L);

    // ---- shared FFN over 64-wide chunks ----
    const long rows64 = BL * H;            // 25600
    gemm(fn, HU, fc_w, HU, fc_b, attn, HU, (int)rows64, HU, HU, 0, stream);
    ffn_act_kernel<<<(int)((rows64 * HU + 255) / 256), 256, 0, stream>>>(attn, fn, rows64 * HU);
    gemm(attn, HU, fc2_w, HU, fc2_b, tmp2, HU, (int)rows64, HU, HU, 0, stream);
    ffn_ln_kernel<<<(int)rows64, 64, 0, stream>>>(fn, tmp2, ffn_ln_g, ffn_ln_b, fn);  // y in place
    // head gating -> log_feats (first output)
    ffn_gate_kernel<<<(int)BL, 64, 0, stream>>>(fn, ffn_gate_w, log_feats);

    // ---- all-item scores (dominant GEMMs) ----
    float* ph = qe;                        // reuse expert-q slab: [NTOT, 256]
    gemm(item_emb, D, lab_w, D, nullptr, ph, D, NTOT, D, D, 0, stream);
    gemm(log_feats, D, ph, D, nullptr, score, (long)NTOT, (int)BL, NTOT, D, 0, stream);
}
